// GATV2Model_12163347382338
// MI455X (gfx1250) — compile-verified, matrix operand order checked by reading, not run on hardware
//
#include <hip/hip_runtime.h>
#include <hip/hip_bf16.h>
#include <cstdint>
#include <cstddef>

// ---------------------------------------------------------------------------
// GATv2 (3 layers) for MI455X / gfx1250.
//  * Node transforms: v_wmma_f32_16x16x32_f16, A and B pre-packed into the
//    ISA 7.12.2 fragment layout so every fragment is one 32B contiguous load.
//  * Edge softmax/aggregation: float4 gathers + native f32/u32 L2 atomics.
//  * Assumes N (nodes) is a multiple of 16 (holds: 50000 = 3125*16).
// ---------------------------------------------------------------------------

typedef __attribute__((ext_vector_type(16))) _Float16 v16h;
typedef __attribute__((ext_vector_type(8)))  float    v8f;

#define GAT_HEADS 12
#define GAT_HID   32
#define GAT_FEAT  (GAT_HEADS * GAT_HID)   // 384
#define GAT_SLOPE 0.2f

// ------------------------- ordered-float encoding for atomicMax ------------
__device__ __forceinline__ unsigned ford_enc(float f) {
  unsigned u = __float_as_uint(f);
  return (u & 0x80000000u) ? ~u : (u | 0x80000000u);
}
__device__ __forceinline__ float ford_dec(unsigned u) {
  return __uint_as_float((u & 0x80000000u) ? (u & 0x7fffffffu) : ~u);
}
__device__ __forceinline__ float lrelu(float v) {
  return v > 0.f ? v : GAT_SLOPE * v;
}

// Fragment index helpers (16-bit A 16x32 layout, ISA 7.12.2):
//   element i of lane L holds K = kb*32 + (i>>3)*16 + ((L>>4)<<3) + (i&7),
//   row/col = L & 15.  Packed linear index: ((tile*KB + kb)*32 + lane)*16 + i.
__device__ __forceinline__ int frag_k(int kb, int lane, int i) {
  return kb * 32 + ((i >> 3) << 4) + ((lane >> 4) << 3) + (i & 7);
}

// ------------------------- utility kernels ---------------------------------
__global__ void k_zero_u32(unsigned* __restrict__ p, long long n) {
  long long t = (long long)blockIdx.x * blockDim.x + threadIdx.x;
  if (t < n) p[t] = 0u;
}

// Pack A: src f32 row-major [M x K] -> f16 fragment layout. M % 16 == 0.
__global__ void k_pack_a(const float* __restrict__ src, _Float16* __restrict__ dst,
                         int M, int K) {
  long long t = (long long)blockIdx.x * blockDim.x + threadIdx.x;
  if (t >= (long long)M * K) return;
  const int KB = K >> 5;
  int i    = (int)(t & 15);
  int lane = (int)((t >> 4) & 31);
  long long blk = t >> 9;                 // (mtile*KB + kb)
  int kb    = (int)(blk % KB);
  int mtile = (int)(blk / KB);
  int row = mtile * 16 + (lane & 15);
  int kk  = frag_k(kb, lane, i);
  dst[t] = (_Float16)src[(size_t)row * K + kk];
}

// Pack B: W f32 row-major [K x 384] -> f16 fragment layout (B 32x16 tiles).
__global__ void k_pack_b(const float* __restrict__ W, _Float16* __restrict__ dst,
                         int K) {
  long long t = (long long)blockIdx.x * blockDim.x + threadIdx.x;
  if (t >= (long long)GAT_FEAT * K) return;
  const int KB = K >> 5;
  int i    = (int)(t & 15);
  int lane = (int)((t >> 4) & 31);
  long long blk = t >> 9;                 // (ntile*KB + kb)
  int kb    = (int)(blk % KB);
  int ntile = (int)(blk / KB);
  int col = ntile * 16 + (lane & 15);
  int kk  = frag_k(kb, lane, i);
  dst[t] = (_Float16)W[(size_t)kk * GAT_FEAT + col];
}

// ------------------------- WMMA GEMM ---------------------------------------
// C[M x 384] = A[M x K] * B[K x 384] + bias.  A/B pre-packed fragments.
// One wave: 1 M-tile x 6 N-tiles (4 waves/block cover all 24 N tiles).
// M % 16 == 0; grid.x = M/16.  EXEC all-ones throughout (WMMA requirement).
template <int K>
__global__ __launch_bounds__(128) void k_wmma_gemm(
    const _Float16* __restrict__ Ap, const _Float16* __restrict__ Bp,
    const float* __restrict__ bias, float* __restrict__ C) {
  constexpr int KB = K / 32;
  constexpr int NT = 6;                    // N tiles per wave
  const int lane   = threadIdx.x & 31;
  const int wave   = threadIdx.x >> 5;
  const int tileM  = blockIdx.x;
  const int tileN0 = wave * NT;

  v8f acc[NT] = {};
  const v16h* apv = (const v16h*)Ap + (size_t)tileM * KB * 32 + lane;
#pragma unroll
  for (int kb = 0; kb < KB; ++kb) {
    v16h av = apv[kb * 32];
#pragma unroll
    for (int t = 0; t < NT; ++t) {
      v16h bv = ((const v16h*)Bp)[((size_t)(tileN0 + t) * KB + kb) * 32 + lane];
      acc[t] = __builtin_amdgcn_wmma_f32_16x16x32_f16(
          false, av, false, bv, (short)0, acc[t], false, false);
    }
  }

  const int l15 = lane & 15, half = lane >> 4;
#pragma unroll
  for (int t = 0; t < NT; ++t) {
    int col = (tileN0 + t) * 16 + l15;
    float bb = bias[col];
#pragma unroll
    for (int r = 0; r < 8; ++r) {
      int row = tileM * 16 + r + (half << 3);       // f32 C/D layout
      C[(size_t)row * GAT_FEAT + col] = acc[t][r] + bb;
    }
  }
}

// ------------------------- edge kernels (HEADS=12, C=32) -------------------
__device__ __forceinline__ void edge_sd(const int* __restrict__ ei, int E,
                                        int e, int& s, int& d) {
  if (e < E) { s = ei[e]; d = ei[E + e]; }
  else       { s = e - E; d = e - E; }     // self-loop
}

__global__ void k_edge_logits(const int* __restrict__ ei, int E, int Etot,
                              const float* __restrict__ XL,
                              const float* __restrict__ XR,
                              const float* __restrict__ att,
                              float* __restrict__ logits,
                              unsigned* __restrict__ mEnc) {
  long long t = (long long)blockIdx.x * blockDim.x + threadIdx.x;
  if (t >= (long long)Etot * GAT_HEADS) return;
  int e = (int)(t / GAT_HEADS);
  int h = (int)(t - (long long)e * GAT_HEADS);
  int s, d; edge_sd(ei, E, e, s, d);
  const float4* xl = (const float4*)(XL + (size_t)s * GAT_FEAT + h * GAT_HID);
  const float4* xr = (const float4*)(XR + (size_t)d * GAT_FEAT + h * GAT_HID);
  const float4* at = (const float4*)(att + h * GAT_HID);
  float acc = 0.f;
#pragma unroll
  for (int c = 0; c < GAT_HID / 4; ++c) {
    float4 a = xl[c], b = xr[c], w = at[c];
    acc += w.x * lrelu(a.x + b.x) + w.y * lrelu(a.y + b.y) +
           w.z * lrelu(a.z + b.z) + w.w * lrelu(a.w + b.w);
  }
  logits[t] = acc;
  atomicMax(mEnc + (size_t)d * GAT_HEADS + h, ford_enc(acc));
}

__global__ void k_edge_exp(const int* __restrict__ ei, int E, int Etot,
                           const unsigned* __restrict__ mEnc,
                           float* __restrict__ logits,
                           float* __restrict__ denom) {
  long long t = (long long)blockIdx.x * blockDim.x + threadIdx.x;
  if (t >= (long long)Etot * GAT_HEADS) return;
  int e = (int)(t / GAT_HEADS);
  int h = (int)(t - (long long)e * GAT_HEADS);
  int s, d; edge_sd(ei, E, e, s, d); (void)s;
  float m = ford_dec(mEnc[(size_t)d * GAT_HEADS + h]);
  float a = expf(logits[t] - m);
  logits[t] = a;
  atomicAdd(denom + (size_t)d * GAT_HEADS + h, a);
}

__global__ void k_edge_scatter(const int* __restrict__ ei, int E, int Etot,
                               const float* __restrict__ XL,
                               const float* __restrict__ logits,
                               const float* __restrict__ denom,
                               float* __restrict__ AGG) {
  long long t = (long long)blockIdx.x * blockDim.x + threadIdx.x;
  if (t >= (long long)Etot * GAT_HEADS) return;
  int e = (int)(t / GAT_HEADS);
  int h = (int)(t - (long long)e * GAT_HEADS);
  int s, d; edge_sd(ei, E, e, s, d);
  float alpha = logits[t] / (denom[(size_t)d * GAT_HEADS + h] + 1e-16f);
  const float4* xl = (const float4*)(XL + (size_t)s * GAT_FEAT + h * GAT_HID);
  float* o = AGG + (size_t)d * GAT_FEAT + h * GAT_HID;
#pragma unroll
  for (int c = 0; c < GAT_HID / 4; ++c) {
    float4 a = xl[c];
    atomicAdd(o + 4 * c + 0, a.x * alpha);
    atomicAdd(o + 4 * c + 1, a.y * alpha);
    atomicAdd(o + 4 * c + 2, a.z * alpha);
    atomicAdd(o + 4 * c + 3, a.w * alpha);
  }
}

// mean over heads + bias + ELU; emit f16 directly in A-fragment layout
// for the next GEMM (K = 32 -> KB = 1).
__global__ void k_finalize_layer(const float* __restrict__ AGG,
                                 const float* __restrict__ bias,
                                 _Float16* __restrict__ H16p, int Nnodes) {
  long long t = (long long)blockIdx.x * blockDim.x + threadIdx.x;
  if (t >= (long long)Nnodes * GAT_HID) return;
  int n = (int)(t >> 5), c = (int)(t & 31);
  float sum = 0.f;
#pragma unroll
  for (int h = 0; h < GAT_HEADS; ++h)
    sum += AGG[(size_t)n * GAT_FEAT + h * GAT_HID + c];
  float v = sum * (1.0f / GAT_HEADS) + bias[c];
  v = v > 0.f ? v : expm1f(v);                     // ELU
  // fragment position for (row=n, k=c), KB=1:
  int lane = (((c >> 3) & 1) << 4) | (n & 15);
  int i    = ((c >> 4) << 3) | (c & 7);
  size_t idx = (((size_t)(n >> 4)) * 32 + lane) * 16 + i;
  H16p[idx] = (_Float16)v;
}

// ------------------------- output layer (heads=1, NC=2) --------------------
__global__ void k_out_transform(const _Float16* __restrict__ H16p,
                                const float* __restrict__ Wl, const float* __restrict__ bl,
                                const float* __restrict__ Wr, const float* __restrict__ br,
                                float* __restrict__ XLo, float* __restrict__ XRo,
                                int Nnodes) {
  int n = blockIdx.x * blockDim.x + threadIdx.x;
  if (n >= Nnodes) return;
  float l0 = bl[0], l1 = bl[1], r0 = br[0], r1 = br[1];
#pragma unroll
  for (int k = 0; k < GAT_HID; ++k) {
    int lane = (((k >> 3) & 1) << 4) | (n & 15);
    int i    = ((k >> 4) << 3) | (k & 7);
    float hv = (float)H16p[(((size_t)(n >> 4)) * 32 + lane) * 16 + i];
    l0 += hv * Wl[k * 2 + 0];  l1 += hv * Wl[k * 2 + 1];
    r0 += hv * Wr[k * 2 + 0];  r1 += hv * Wr[k * 2 + 1];
  }
  XLo[(size_t)n * 2 + 0] = l0;  XLo[(size_t)n * 2 + 1] = l1;
  XRo[(size_t)n * 2 + 0] = r0;  XRo[(size_t)n * 2 + 1] = r1;
}

__global__ void k_out_logits(const int* __restrict__ ei, int E, int Etot,
                             const float* __restrict__ XLo,
                             const float* __restrict__ XRo,
                             const float* __restrict__ atto,
                             float* __restrict__ logits,
                             unsigned* __restrict__ mEnc) {
  int e = blockIdx.x * blockDim.x + threadIdx.x;
  if (e >= Etot) return;
  int s, d; edge_sd(ei, E, e, s, d);
  float l = atto[0] * lrelu(XLo[(size_t)s * 2 + 0] + XRo[(size_t)d * 2 + 0]) +
            atto[1] * lrelu(XLo[(size_t)s * 2 + 1] + XRo[(size_t)d * 2 + 1]);
  logits[e] = l;
  atomicMax(mEnc + d, ford_enc(l));
}

__global__ void k_out_exp(const int* __restrict__ ei, int E, int Etot,
                          const unsigned* __restrict__ mEnc,
                          float* __restrict__ logits,
                          float* __restrict__ denom) {
  int e = blockIdx.x * blockDim.x + threadIdx.x;
  if (e >= Etot) return;
  int s, d; edge_sd(ei, E, e, s, d); (void)s;
  float a = expf(logits[e] - ford_dec(mEnc[d]));
  logits[e] = a;
  atomicAdd(denom + d, a);
}

__global__ void k_out_scatter(const int* __restrict__ ei, int E, int Etot,
                              const float* __restrict__ XLo,
                              const float* __restrict__ logits,
                              const float* __restrict__ denom,
                              float* __restrict__ AGGo) {
  int e = blockIdx.x * blockDim.x + threadIdx.x;
  if (e >= Etot) return;
  int s, d; edge_sd(ei, E, e, s, d);
  float alpha = logits[e] / (denom[d] + 1e-16f);
  atomicAdd(AGGo + (size_t)d * 2 + 0, XLo[(size_t)s * 2 + 0] * alpha);
  atomicAdd(AGGo + (size_t)d * 2 + 1, XLo[(size_t)s * 2 + 1] * alpha);
}

__global__ void k_out_final(const float* __restrict__ AGGo,
                            const float* __restrict__ biaso,
                            float* __restrict__ out, int Nnodes) {
  int t = blockIdx.x * blockDim.x + threadIdx.x;
  if (t >= Nnodes * 2) return;
  out[t] = AGGo[t] + biaso[t & 1];
}

// ---------------------------------------------------------------------------
extern "C" void kernel_launch(void* const* d_in, const int* in_sizes, int n_in,
                              void* d_out, int out_size, void* d_ws, size_t ws_size,
                              hipStream_t stream) {
  (void)n_in; (void)out_size; (void)ws_size;
  const float* x     = (const float*)d_in[0];
  const int*   ei    = (const int*)  d_in[1];
  const float* Wl0   = (const float*)d_in[2];
  const float* bl0   = (const float*)d_in[3];
  const float* Wr0   = (const float*)d_in[4];
  const float* br0   = (const float*)d_in[5];
  const float* att0  = (const float*)d_in[6];
  const float* bias0 = (const float*)d_in[7];
  const float* Wl1   = (const float*)d_in[8];
  const float* bl1   = (const float*)d_in[9];
  const float* Wr1   = (const float*)d_in[10];
  const float* br1   = (const float*)d_in[11];
  const float* att1  = (const float*)d_in[12];
  const float* bias1 = (const float*)d_in[13];
  const float* Wlo   = (const float*)d_in[14];
  const float* blo   = (const float*)d_in[15];
  const float* Wro   = (const float*)d_in[16];
  const float* bro   = (const float*)d_in[17];
  const float* atto  = (const float*)d_in[18];
  const float* biaso = (const float*)d_in[19];
  float* out = (float*)d_out;

  const int IN   = 128;
  const int Nn   = in_sizes[0] / IN;     // 50000 (multiple of 16)
  const int E    = in_sizes[1] / 2;      // 400000
  const int Etot = E + Nn;               // 450000

  // ----- workspace carve-out (256B aligned) -----
  char* wp = (char*)d_ws;
  auto alloc = [&](size_t bytes) -> void* {
    void* p = (void*)wp;
    wp += (bytes + 255) & ~(size_t)255;
    return p;
  };
  _Float16* Ap    = (_Float16*)alloc((size_t)Nn * IN * 2);          // packed A (layer0)
  _Float16* Bl16  = (_Float16*)alloc((size_t)IN * GAT_FEAT * 2);    // packed B (Wl)
  _Float16* Br16  = (_Float16*)alloc((size_t)IN * GAT_FEAT * 2);    // packed B (Wr)
  float*    XL    = (float*)   alloc((size_t)Nn * GAT_FEAT * 4);
  float*    XR    = (float*)   alloc((size_t)Nn * GAT_FEAT * 4);
  float*    AGG   = (float*)   alloc((size_t)Nn * GAT_FEAT * 4);
  float*    elog  = (float*)   alloc((size_t)Etot * GAT_HEADS * 4);
  unsigned* mEnc  = (unsigned*)alloc((size_t)Nn * GAT_HEADS * 4);
  float*    denom = (float*)   alloc((size_t)Nn * GAT_HEADS * 4);
  _Float16* H16p  = (_Float16*)alloc((size_t)Nn * GAT_HID * 2);     // packed A (layer1)
  float*    XLo   = (float*)   alloc((size_t)Nn * 2 * 4);
  float*    XRo   = (float*)   alloc((size_t)Nn * 2 * 4);
  float*    AGGo  = (float*)   alloc((size_t)Nn * 2 * 4);
  unsigned* mEo   = (unsigned*)alloc((size_t)Nn * 4);
  float*    deno  = (float*)   alloc((size_t)Nn * 4);
  float*    elogo = (float*)   alloc((size_t)Etot * 4);

  const int B = 256;
  auto blocks = [](long long n, int b) { return (unsigned)((n + b - 1) / b); };
  auto zero = [&](void* p, long long n_u32) {
    k_zero_u32<<<blocks(n_u32, B), B, 0, stream>>>((unsigned*)p, n_u32);
  };

  // ----- edge phase helper (12 heads) -----
  auto edge_phase = [&](const float* att) {
    long long nt = (long long)Etot * GAT_HEADS;
    zero(mEnc,  (long long)Nn * GAT_HEADS);
    zero(denom, (long long)Nn * GAT_HEADS);
    zero(AGG,   (long long)Nn * GAT_FEAT);
    k_edge_logits <<<blocks(nt, B), B, 0, stream>>>(ei, E, Etot, XL, XR, att, elog, mEnc);
    k_edge_exp    <<<blocks(nt, B), B, 0, stream>>>(ei, E, Etot, mEnc, elog, denom);
    k_edge_scatter<<<blocks(nt, B), B, 0, stream>>>(ei, E, Etot, XL, elog, denom, AGG);
  };

  // ================= layer 0 (K = 128) =================
  k_pack_a<<<blocks((long long)Nn * IN, B), B, 0, stream>>>(x, Ap, Nn, IN);
  k_pack_b<<<blocks((long long)IN * GAT_FEAT, B), B, 0, stream>>>(Wl0, Bl16, IN);
  k_pack_b<<<blocks((long long)IN * GAT_FEAT, B), B, 0, stream>>>(Wr0, Br16, IN);
  k_wmma_gemm<128><<<dim3(Nn / 16, 1, 1), 128, 0, stream>>>(Ap, Bl16, bl0, XL);
  k_wmma_gemm<128><<<dim3(Nn / 16, 1, 1), 128, 0, stream>>>(Ap, Br16, br0, XR);
  edge_phase(att0);
  k_finalize_layer<<<blocks((long long)Nn * GAT_HID, B), B, 0, stream>>>(AGG, bias0, H16p, Nn);

  // ================= layer 1 (K = 32) =================
  k_pack_b<<<blocks((long long)GAT_HID * GAT_FEAT, B), B, 0, stream>>>(Wl1, Bl16, GAT_HID);
  k_pack_b<<<blocks((long long)GAT_HID * GAT_FEAT, B), B, 0, stream>>>(Wr1, Br16, GAT_HID);
  k_wmma_gemm<32><<<dim3(Nn / 16, 1, 1), 128, 0, stream>>>(H16p, Bl16, bl1, XL);
  k_wmma_gemm<32><<<dim3(Nn / 16, 1, 1), 128, 0, stream>>>(H16p, Br16, br1, XR);
  edge_phase(att1);
  k_finalize_layer<<<blocks((long long)Nn * GAT_HID, B), B, 0, stream>>>(AGG, bias1, H16p, Nn);

  // ================= output layer (heads=1, NC=2) =================
  zero(mEo,  Nn);
  zero(deno, Nn);
  zero(AGGo, (long long)Nn * 2);
  k_out_transform<<<blocks(Nn, B), B, 0, stream>>>(H16p, Wlo, blo, Wro, bro, XLo, XRo, Nn);
  k_out_logits  <<<blocks(Etot, B), B, 0, stream>>>(ei, E, Etot, XLo, XRo, atto, elogo, mEo);
  k_out_exp     <<<blocks(Etot, B), B, 0, stream>>>(ei, E, Etot, mEo, elogo, deno);
  k_out_scatter <<<blocks(Etot, B), B, 0, stream>>>(ei, E, Etot, XLo, elogo, deno, AGGo);
  k_out_final   <<<blocks((long long)Nn * 2, B), B, 0, stream>>>(AGGo, biaso, out, Nn);
}
